// BiattGRU_58222576664715
// MI455X (gfx1250) — matrix operand order repaired and checked by reading, not compiled, not run on hardware
//
#include <hip/hip_runtime.h>
#include <hip/hip_bf16.h>
#include <stdint.h>

#define BB 64
#define TT 2048
#define DD 200
#define HH 100
#define CC 8
#define MROWS (BB * TT)   // 131072
#define GSTRIDE 384       // per-direction gi stride: 3 gates * 128 (pow2 pads)
#define OSTRIDE 208       // out row stride (2H=200 padded to 16-mult)

typedef __attribute__((ext_vector_type(16))) __bf16 v16bf;
typedef __attribute__((ext_vector_type(8)))  __bf16 v8bf;
typedef __attribute__((ext_vector_type(8)))  float  v8f;

// native RNE converts (gfx1250 has hw bf16 cvt paths)
static __device__ __forceinline__ uint32_t cvt2(float a, float b) {
  union { __bf16 h[2]; uint32_t u; } o;
  o.h[0] = (__bf16)a; o.h[1] = (__bf16)b;
  return o.u;
}
static __device__ __forceinline__ float bf2f(__bf16 h) { return (float)h; }
static __device__ __forceinline__ float sigmoidf_(float x) {
  return 1.0f / (1.0f + __expf(-x));
}
static __device__ __forceinline__ int imin(int a, int b) { return a < b ? a : b; }

// A fragment: 16x32 bf16 tile, row-major, leading dim ldk (k contiguous).
// lane<16: m=lane, k in {0..7, 16..23}; lane>=16: k in {8..15, 24..31}.
static __device__ __forceinline__ v16bf load_a(const __bf16* tile, int ldk, int lane) {
  const __bf16* p = tile + (lane & 15) * ldk + ((lane & 16) ? 8 : 0);
  v8bf lo = *(const v8bf*)(p);
  v8bf hi = *(const v8bf*)(p + 16);
  v16bf r;
#pragma unroll
  for (int j = 0; j < 8; ++j) { r[j] = lo[j]; r[j + 8] = hi[j]; }
  return r;
}
// B fragment: 32x16 bf16 stored K-major: tile[n*ldk + k]. lane n=lane&15,
// lane<16: k in {0..7, 16..23}; lane>=16: k in {8..15, 24..31}.
static __device__ __forceinline__ v16bf load_b(const __bf16* tile, int ldk, int lane) {
  const __bf16* p = tile + (lane & 15) * ldk + ((lane & 16) ? 8 : 0);
  v8bf lo = *(const v8bf*)(p);
  v8bf hi = *(const v8bf*)(p + 16);
  v16bf r;
#pragma unroll
  for (int j = 0; j < 8; ++j) { r[j] = lo[j]; r[j + 8] = hi[j]; }
  return r;
}
#define WMMA_BF16(A, Bm, C) \
  __builtin_amdgcn_wmma_f32_16x16x32_bf16(false, (A), false, (Bm), (short)0, (C), false, false)

// ---------------------------------------------------------------------------
// K1: gi = x @ wih^T, both directions fused. N space [0,768): dir = n>=384,
// per-dir layout [r:0..127][z:128..255][n:256..383] (first 100 of each
// 128-block meaningful). Dir boundary tile-aligned -> uniform selects.
// ---------------------------------------------------------------------------
__global__ void __launch_bounds__(128) k_gi(
    const float* __restrict__ x, const float* __restrict__ wihF,
    const float* __restrict__ wihB, __bf16* __restrict__ giF,
    __bf16* __restrict__ giB) {
  __shared__ alignas(16) __bf16 As[64 * 32];   // [r][k]
  __shared__ alignas(16) __bf16 Bs[32 * 32];   // [n][k] K-major
  const int m0 = blockIdx.x * 64;
  const int n0 = blockIdx.y * 32;                     // [0, 768)
  const int isb = (n0 >= GSTRIDE);                    // block-uniform
  const int nbase = n0 - (isb ? GSTRIDE : 0);
  const float* __restrict__ wih = isb ? wihB : wihF;
  __bf16* __restrict__ gout = isb ? giB : giF;
  const int tid = threadIdx.x, wave = tid >> 5, lane = tid & 31;
  v8f acc0 = {0,0,0,0,0,0,0,0}, acc1 = {0,0,0,0,0,0,0,0};
  for (int k0 = 0; k0 < 224; k0 += 32) {
    // A: 64x32 bf16 = 1024 u32; even dims -> per-pair mask, b64 loads
#pragma unroll
    for (int i = tid; i < 1024; i += 128) {
      int r = i >> 4, c0 = k0 + (i & 15) * 2;
      const float2* px = (const float2*)(x + (size_t)(m0 + r) * DD);
      float2 v = px[imin(c0, DD - 2) >> 1];
      float msk = (c0 < DD) ? 1.f : 0.f;
      ((uint32_t*)As)[i] = cvt2(v.x * msk, v.y * msk);
    }
    // B: 32x32 bf16 = 512 u32
#pragma unroll
    for (int i = tid; i < 512; i += 128) {
      int n = i >> 4, c0 = k0 + (i & 15) * 2;
      int nl = nbase + n;
      int g = nl >> 7, jj = nl & 127;
      const float2* pw =
          (const float2*)(wih + (size_t)(g * HH + imin(jj, HH - 1)) * DD);
      float2 v = pw[imin(c0, DD - 2) >> 1];
      float msk = ((jj < HH) && (c0 < DD)) ? 1.f : 0.f;
      ((uint32_t*)Bs)[i] = cvt2(v.x * msk, v.y * msk);
    }
    __syncthreads();
    v16bf a  = load_a(As + wave * 16 * 32, 32, lane);
    v16bf b0 = load_b(Bs, 32, lane);
    v16bf b1 = load_b(Bs + 16 * 32, 32, lane);
    acc0 = WMMA_BF16(a, b0, acc0);
    acc1 = WMMA_BF16(a, b1, acc1);
    __syncthreads();
  }
  const int hi8 = (lane & 16) ? 8 : 0, nl16 = lane & 15;
#pragma unroll
  for (int v = 0; v < 8; ++v) {
    size_t row = (size_t)(m0 + wave * 16 + v + hi8);
#pragma unroll
    for (int nt = 0; nt < 2; ++nt) {
      gout[row * GSTRIDE + nbase + nt * 16 + nl16] =
          (__bf16)(nt ? acc1[v] : acc0[v]);
    }
  }
}

// ---------------------------------------------------------------------------
// K2: recurrence. 8 persistent WGs = 2 dirs x 4 batch-tiles(16). whh resident
// in LDS (K-major bf16, gate stride 128 rows). 7 waves own one 16-wide j-tile
// each and compute r/z/n gate tiles together; gate math + h-update in regs.
// ---------------------------------------------------------------------------
__global__ void __launch_bounds__(256) k_gru(
    const float* __restrict__ whhF, const float* __restrict__ whhB,
    const float* __restrict__ bihF, const float* __restrict__ bhhF,
    const float* __restrict__ bihB, const float* __restrict__ bhhB,
    const __bf16* __restrict__ giF, const __bf16* __restrict__ giB,
    float* __restrict__ outb) {
  __shared__ alignas(16) __bf16 Bw[384 * 128];   // 96KB, [n][k] K-major
  __shared__ float hF[16 * 112];                 // fp32 hidden state
  __shared__ alignas(16) __bf16 hA[16 * 128];    // bf16 A-stage (K padded)
  const int dir = blockIdx.x >> 2;
  const int bt  = blockIdx.x & 3;
  const float* whh = dir ? whhB : whhF;
  const float* bih = dir ? bihB : bihF;
  const float* bhh = dir ? bhhB : bhhF;
  const __bf16* gi = dir ? giB : giF;
  const int tid = threadIdx.x, wave = tid >> 5, lane = tid & 31;

  // Bw[n][k] = whh[row(n)][k]; 384*128 bf16 = 24576 u32, 96/thread
  for (int i = tid; i < 384 * 64; i += 256) {
    int n = i >> 6, kp = (i & 63) * 2;
    int g = n >> 7, jj = n & 127;
    const float2* pw =
        (const float2*)(whh + (size_t)(g * HH + imin(jj, HH - 1)) * HH);
    float2 v = pw[imin(kp, HH - 2) >> 1];
    float msk = ((jj < HH) && (kp < HH)) ? 1.f : 0.f;
    ((uint32_t*)Bw)[i] = cvt2(v.x * msk, v.y * msk);
  }
#pragma unroll
  for (int i = tid; i < 16 * 112; i += 256) hF[i] = 0.f;
  __syncthreads();

  const int jt = wave;                        // 0..6 compute, 7 = helper
  const int j = jt * 16 + (lane & 15);
  const int hi8 = (lane & 16) ? 8 : 0;
  float br_ = 0, bz_ = 0, bn_ = 0, cr_ = 0, cz_ = 0, cn_ = 0;
  if (j < HH) {
    br_ = bih[j]; bz_ = bih[HH + j]; bn_ = bih[2 * HH + j];
    cr_ = bhh[j]; cz_ = bhh[HH + j]; cn_ = bhh[2 * HH + j];
  }

  for (int t = 0; t < TT; ++t) {
    const int ttv = dir ? (TT - 1 - t) : t;
    // hA rebuild: 1024 u32, exactly 4 iters/thread, pair-masked b64 loads
#pragma unroll
    for (int i = tid; i < 1024; i += 256) {
      int b = i >> 6, kp = (i & 63) * 2;
      const float2* ph = (const float2*)(hF + b * 112);
      float2 v = ph[imin(kp, 110) >> 1];
      float msk = (kp < HH) ? 1.f : 0.f;
      ((uint32_t*)hA)[i] = cvt2(v.x * msk, v.y * msk);
    }
    __syncthreads();
    if (wave < 7) {            // wave-uniform branch; EXEC all-ones inside
      v8f aR = {0,0,0,0,0,0,0,0}, aZ = {0,0,0,0,0,0,0,0}, aN = {0,0,0,0,0,0,0,0};
#pragma unroll
      for (int kk = 0; kk < 128; kk += 32) {
        v16bf a   = load_a(hA + kk, 128, lane);
        v16bf brf = load_b(Bw + (size_t)(0   + jt * 16) * 128 + kk, 128, lane);
        v16bf bzf = load_b(Bw + (size_t)(128 + jt * 16) * 128 + kk, 128, lane);
        v16bf bnf = load_b(Bw + (size_t)(256 + jt * 16) * 128 + kk, 128, lane);
        aR = WMMA_BF16(a, brf, aR);
        aZ = WMMA_BF16(a, bzf, aZ);
        aN = WMMA_BF16(a, bnf, aN);
      }
      if (j < HH) {
#pragma unroll
        for (int v = 0; v < 8; ++v) {
          int bl = v + hi8;                                 // local batch row
          size_t grow = (size_t)(bt * 16 + bl) * TT + ttv;  // global (b,t) row
          float giR = bf2f(gi[grow * GSTRIDE + j]);
          float giZ = bf2f(gi[grow * GSTRIDE + 128 + j]);
          float giN = bf2f(gi[grow * GSTRIDE + 256 + j]);
          float r  = sigmoidf_(giR + br_ + aR[v] + cr_);
          float z  = sigmoidf_(giZ + bz_ + aZ[v] + cz_);
          float nn = tanhf(giN + bn_ + r * (aN[v] + cn_));
          float hold = hF[bl * 112 + j];
          float hnew = (1.f - z) * nn + z * hold;
          hF[bl * 112 + j] = hnew;
          outb[grow * OSTRIDE + dir * HH + j] = hnew;
        }
      }
    }
    __syncthreads();
  }
}

// ---------------------------------------------------------------------------
// K3a: U = out @ attu_w^T + attu_b  (131072 x 208, K=200 pad 224)
// ---------------------------------------------------------------------------
__global__ void __launch_bounds__(128) k_ugemm(
    const float* __restrict__ outb, const float* __restrict__ attu_w,
    const float* __restrict__ attu_b, float* __restrict__ U) {
  __shared__ alignas(16) __bf16 As[64 * 32];
  __shared__ alignas(16) __bf16 Bs[32 * 32];
  const int m0 = blockIdx.x * 64;
  const int n0 = blockIdx.y * 32;   // [0,224)
  const int tid = threadIdx.x, wave = tid >> 5, lane = tid & 31;
  v8f acc0 = {0,0,0,0,0,0,0,0}, acc1 = {0,0,0,0,0,0,0,0};
  for (int k0 = 0; k0 < 224; k0 += 32) {
#pragma unroll
    for (int i = tid; i < 1024; i += 128) {
      int r = i >> 4, c0 = k0 + (i & 15) * 2;
      const float2* po = (const float2*)(outb + (size_t)(m0 + r) * OSTRIDE);
      float2 v = po[imin(c0, 2 * HH - 2) >> 1];
      float msk = (c0 < 2 * HH) ? 1.f : 0.f;
      ((uint32_t*)As)[i] = cvt2(v.x * msk, v.y * msk);
    }
#pragma unroll
    for (int i = tid; i < 512; i += 128) {
      int n = i >> 4, c0 = k0 + (i & 15) * 2;
      int ng = n0 + n;
      const float2* pw =
          (const float2*)(attu_w + (size_t)imin(ng, 2 * HH - 1) * (2 * HH));
      float2 v = pw[imin(c0, 2 * HH - 2) >> 1];
      float msk = ((ng < 2 * HH) && (c0 < 2 * HH)) ? 1.f : 0.f;
      ((uint32_t*)Bs)[i] = cvt2(v.x * msk, v.y * msk);
    }
    __syncthreads();
    v16bf a  = load_a(As + wave * 16 * 32, 32, lane);
    v16bf b0 = load_b(Bs, 32, lane);
    v16bf b1 = load_b(Bs + 16 * 32, 32, lane);
    acc0 = WMMA_BF16(a, b0, acc0);
    acc1 = WMMA_BF16(a, b1, acc1);
    __syncthreads();
  }
  const int hi8 = (lane & 16) ? 8 : 0, nl16 = lane & 15;
#pragma unroll
  for (int v = 0; v < 8; ++v) {
    size_t row = (size_t)(m0 + wave * 16 + v + hi8);
#pragma unroll
    for (int nt = 0; nt < 2; ++nt) {
      int ng = n0 + nt * 16 + nl16;
      if (ng < OSTRIDE) {
        float val = (nt ? acc1[v] : acc0[v]) + (ng < 2 * HH ? attu_b[ng] : 0.f);
        U[row * OSTRIDE + ng] = val;
      }
    }
  }
}

// ---------------------------------------------------------------------------
// K3b: per-t batch-norm stats over B, tanh, score = <.,atts_w>
// ---------------------------------------------------------------------------
__global__ void __launch_bounds__(256) k_bnscore(
    const float* __restrict__ U, const float* __restrict__ bn_g,
    const float* __restrict__ bn_b, const float* __restrict__ atts_w,
    float* __restrict__ scores) {
  __shared__ float mu[224], rs[224], gg[224], bb[224], aw[224];
  const int t = blockIdx.x, tid = threadIdx.x;
  if (tid < 2 * HH) {
    float s = 0.f, s2 = 0.f;
    for (int b = 0; b < BB; ++b) {
      float v = U[((size_t)b * TT + t) * OSTRIDE + tid];
      s += v; s2 += v * v;
    }
    float m = s * (1.f / BB);
    float var = s2 * (1.f / BB) - m * m;   // biased, as BN
    mu[tid] = m;
    rs[tid] = rsqrtf(var + 1e-5f);
    gg[tid] = bn_g[tid]; bb[tid] = bn_b[tid]; aw[tid] = atts_w[tid];
  }
  __syncthreads();
  const int wave = tid >> 5, lane = tid & 31;
  for (int b = wave; b < BB; b += 8) {
    float acc = 0.f;
    for (int e = lane; e < 2 * HH; e += 32) {
      float v = U[((size_t)b * TT + t) * OSTRIDE + e];
      float uu = tanhf((v - mu[e]) * rs[e] * gg[e] + bb[e]);
      acc += uu * aw[e];
    }
#pragma unroll
    for (int off = 16; off > 0; off >>= 1) acc += __shfl_xor(acc, off, 32);
    if (lane == 0) scores[(size_t)b * TT + t] = acc;
  }
}

// ---------------------------------------------------------------------------
// K4: softmax over T, context, classifier
// ---------------------------------------------------------------------------
__global__ void __launch_bounds__(256) k_final(
    const float* __restrict__ outb, const float* __restrict__ scores,
    const float* __restrict__ fc_w, const float* __restrict__ fc_b,
    float* __restrict__ logits) {
  __shared__ float red[256];
  __shared__ float alpha[TT];
  __shared__ float ctx[224];
  const int b = blockIdx.x, tid = threadIdx.x;
  float m = -3.4e38f;
  for (int t = tid; t < TT; t += 256) m = fmaxf(m, scores[(size_t)b * TT + t]);
  red[tid] = m; __syncthreads();
  for (int s = 128; s > 0; s >>= 1) {
    if (tid < s) red[tid] = fmaxf(red[tid], red[tid + s]);
    __syncthreads();
  }
  m = red[0];
  __syncthreads();
  float ssum = 0.f;
  for (int t = tid; t < TT; t += 256) {
    float e = __expf(scores[(size_t)b * TT + t] - m);
    alpha[t] = e;
    ssum += e;
  }
  red[tid] = ssum; __syncthreads();
  for (int s = 128; s > 0; s >>= 1) {
    if (tid < s) red[tid] += red[tid + s];
    __syncthreads();
  }
  const float inv = 1.f / red[0];
  __syncthreads();
  if (tid < 2 * HH) {
    float acc = 0.f;
    for (int t = 0; t < TT; ++t) {
      __builtin_prefetch(outb + ((size_t)b * TT + t + 8) * OSTRIDE + tid, 0, 0);
      acc += outb[((size_t)b * TT + t) * OSTRIDE + tid] * alpha[t];
    }
    ctx[tid] = acc * inv;
  }
  __syncthreads();
  if (tid < CC) {
    float acc = fc_b[tid];
    for (int e = 0; e < 2 * HH; ++e) acc += ctx[e] * fc_w[tid * 2 * HH + e];
    logits[b * CC + tid] = acc;
  }
}

// ---------------------------------------------------------------------------
extern "C" void kernel_launch(void* const* d_in, const int* in_sizes, int n_in,
                              void* d_out, int out_size, void* d_ws, size_t ws_size,
                              hipStream_t stream) {
  (void)in_sizes; (void)n_in; (void)out_size; (void)ws_size;
  const float* x      = (const float*)d_in[0];
  const float* wih_f  = (const float*)d_in[1];
  const float* whh_f  = (const float*)d_in[2];
  const float* bih_f  = (const float*)d_in[3];
  const float* bhh_f  = (const float*)d_in[4];
  const float* wih_b  = (const float*)d_in[5];
  const float* whh_b  = (const float*)d_in[6];
  const float* bih_b  = (const float*)d_in[7];
  const float* bhh_b  = (const float*)d_in[8];
  const float* attu_w = (const float*)d_in[9];
  const float* attu_b = (const float*)d_in[10];
  const float* bn_g   = (const float*)d_in[11];
  const float* bn_b   = (const float*)d_in[12];
  const float* atts_w = (const float*)d_in[13];
  const float* fc_w   = (const float*)d_in[14];
  const float* fc_b   = (const float*)d_in[15];

  char* ws = (char*)d_ws;
  const size_t GI_ELEMS  = (size_t)MROWS * GSTRIDE;   // bf16 elems per dir
  const size_t OUT_ELEMS = (size_t)MROWS * OSTRIDE;   // fp32 elems
  __bf16* giF = (__bf16*)ws;
  __bf16* giB = giF + GI_ELEMS;
  float*  outb = (float*)(ws + 2 * GI_ELEMS * sizeof(__bf16));
  float*  U    = (float*)ws;   // reuses gi region after k_gru consumed it
  float*  scores = (float*)(ws + 2 * GI_ELEMS * sizeof(__bf16)
                               + OUT_ELEMS * sizeof(float));
  // ws use: 201.3MB (gi) + 109.1MB (out) + 0.5MB (scores) ~= 311MB

  k_gi<<<dim3(MROWS / 64, (2 * GSTRIDE) / 32), 128, 0, stream>>>(
      x, wih_f, wih_b, giF, giB);
  k_gru<<<8, 256, 0, stream>>>(
      whh_f, whh_b, bih_f, bhh_f, bih_b, bhh_b, giF, giB, outb);
  k_ugemm<<<dim3(MROWS / 64, 224 / 32), 128, 0, stream>>>(
      outb, attu_w, attu_b, U);
  k_bnscore<<<TT, 256, 0, stream>>>(U, bn_g, bn_b, atts_w, scores);
  k_final<<<BB, 256, 0, stream>>>(outb, scores, fc_w, fc_b, (float*)d_out);
}
